// EncoderLayer_30013231465025
// MI455X (gfx1250) — compile-verified, matrix operand order checked by reading, not run on hardware
//
#include <hip/hip_runtime.h>

typedef _Float16 half_t;
typedef __attribute__((ext_vector_type(16))) _Float16 v16h;
typedef __attribute__((ext_vector_type(8)))  _Float16 v8h;
typedef __attribute__((ext_vector_type(8)))  float    v8f;

// ---------- constants (reference dims, all hardcoded) ----------
#define BB   2
#define SS   2048
#define DD   1024
#define HH   16
#define DKV  64
#define DFF_ 4096
#define MROWS (BB*SS)   // 4096

// LDS tile row stride in halfs (32 data + 8 pad -> 80B row stride,
// 20-dword bank stride covers all 16 lanes of a b128 frag read conflict-free)
#define LDS_STRIDE 40

// ---------- helpers ----------
__device__ inline v16h ld16(const half_t* __restrict__ p0, const half_t* __restrict__ p1) {
    v8h a = *(const v8h*)p0;
    v8h b = *(const v8h*)p1;
    v16h r;
#pragma unroll
    for (int i = 0; i < 8; ++i) { r[i] = a[i]; r[i + 8] = b[i]; }
    return r;
}

__device__ inline v16h ld16_lds(const _Float16* p0, const _Float16* p1) {
    v8h a = *(const v8h*)p0;
    v8h b = *(const v8h*)p1;
    v16h r;
#pragma unroll
    for (int i = 0; i < 8; ++i) { r[i] = a[i]; r[i + 8] = b[i]; }
    return r;
}

__device__ inline v8f wmma16(v16h a, v16h b, v8f c) {
    return __builtin_amdgcn_wmma_f32_16x16x32_f16(
        /*neg_a=*/false, a, /*neg_b=*/false, b,
        /*c_mod=*/(short)0, c, /*reuse_a=*/false, /*reuse_b=*/false);
}

__device__ inline float redmax16(float v) {
    v = fmaxf(v, __shfl_xor(v, 1, 32));
    v = fmaxf(v, __shfl_xor(v, 2, 32));
    v = fmaxf(v, __shfl_xor(v, 4, 32));
    v = fmaxf(v, __shfl_xor(v, 8, 32));
    return v;
}
__device__ inline float redsum16(float v) {
    v += __shfl_xor(v, 1, 32);
    v += __shfl_xor(v, 2, 32);
    v += __shfl_xor(v, 4, 32);
    v += __shfl_xor(v, 8, 32);
    return v;
}

// ---------- weight transpose + f32->f16 cast: wt[n][k] = w[k][n] ----------
__global__ __launch_bounds__(256)
void transpose_cast_kernel(const float* __restrict__ w, half_t* __restrict__ wt,
                           int K, int N) {
    size_t idx = (size_t)blockIdx.x * 256 + threadIdx.x;
    if (idx >= (size_t)K * N) return;
    int n = (int)(idx / K);
    int k = (int)(idx % K);
    wt[idx] = (half_t)w[(size_t)k * N + n];
}

// ---------- layernorm over D=1024, f32 in -> f16 out ----------
__global__ __launch_bounds__(256)
void layernorm_f16_kernel(const float* __restrict__ x, const float* __restrict__ g,
                          const float* __restrict__ b, half_t* __restrict__ out) {
    __shared__ float rs[8], rss[8];
    const int row = blockIdx.x;
    const float* xr = x + (size_t)row * DD;
    float v[4];
    float s = 0.f, ss = 0.f;
#pragma unroll
    for (int i = 0; i < 4; ++i) {
        v[i] = xr[threadIdx.x + i * 256];
        s  += v[i];
        ss += v[i] * v[i];
    }
#pragma unroll
    for (int m = 1; m < 32; m <<= 1) {
        s  += __shfl_xor(s,  m, 32);
        ss += __shfl_xor(ss, m, 32);
    }
    const int w = threadIdx.x >> 5;
    if ((threadIdx.x & 31) == 0) { rs[w] = s; rss[w] = ss; }
    __syncthreads();
    float S = 0.f, SSq = 0.f;
#pragma unroll
    for (int i = 0; i < 8; ++i) { S += rs[i]; SSq += rss[i]; }
    const float mu   = S * (1.f / DD);
    const float var  = SSq * (1.f / DD) - mu * mu;
    const float rstd = rsqrtf(var + 1e-5f);
    half_t* orow = out + (size_t)row * DD;
#pragma unroll
    for (int i = 0; i < 4; ++i) {
        int c = threadIdx.x + i * 256;
        orow[c] = (half_t)((v[i] - mu) * rstd * g[c] + b[c]);
    }
}

// ---------- WMMA GEMM, LDS-staged, block tile 128x128, k-step 32 ----------
// C[M,N] = A[M,K] (f16, row-major) x Bt[N,K] (f16, pre-transposed)
// 256 threads = 8 waves = 4(M) x 2(N); wave tile 32x64 -> 8 WMMAs / k-step.
// MODE 0: store f16 row-major
// MODE 1: store f16 to Vt layout [(b*H+h)*DV+dv][S]
// MODE 2: f32 out = acc (+bias[n] if bias) + res[m,n]
// MODE 3: f16 out = relu(acc + bias[n])
template<int MODE>
__global__ __launch_bounds__(256)
void gemm_wmma(const half_t* __restrict__ A, const half_t* __restrict__ Bt,
               const float* __restrict__ bias, const float* __restrict__ res,
               void* __restrict__ outp, int M, int N, int K) {
    __shared__ _Float16 lsA[128 * LDS_STRIDE];   // 10 KB
    __shared__ _Float16 lsB[128 * LDS_STRIDE];   // 10 KB

    const int tid  = threadIdx.x;
    const int lane = tid & 31;
    const int wave = tid >> 5;
    const int wr   = wave & 3;     // wave row: 4 waves along M (32 rows each)
    const int wc   = wave >> 2;    // wave col: 2 waves along N (64 cols each)
    const int col  = lane & 15;
    const int hs   = lane >> 4;    // half-wave select
    const int kb   = hs * 8;

    const int m0 = blockIdx.y * 128;
    const int n0 = blockIdx.x * 128;

    // ---- global->LDS staging map: each thread owns 2 A-chunks + 2 B-chunks (8 halfs each)
    const int trow = tid >> 2;     // 0..63
    const int tseg = tid & 3;      // 0..3 (which 8-half segment of the 32-deep slice)
    const half_t* gA0 = A  + (size_t)(m0 + trow) * K + tseg * 8;
    const half_t* gA1 = gA0 + (size_t)64 * K;
    const half_t* gB0 = Bt + (size_t)(n0 + trow) * K + tseg * 8;
    const half_t* gB1 = gB0 + (size_t)64 * K;
    _Float16* sA0 = &lsA[trow * LDS_STRIDE + tseg * 8];
    _Float16* sA1 = sA0 + 64 * LDS_STRIDE;
    _Float16* sB0 = &lsB[trow * LDS_STRIDE + tseg * 8];
    _Float16* sB1 = sB0 + 64 * LDS_STRIDE;

    // ---- wave fragment read pointers (LDS)
    const _Float16* fA0 = &lsA[(wr * 32 + col) * LDS_STRIDE + kb];
    const _Float16* fA1 = fA0 + 16 * LDS_STRIDE;
    const _Float16* fB0 = &lsB[(wc * 64 + col) * LDS_STRIDE + kb];

    v8f c[2][4] = {};

    // prologue: fetch tile 0 into registers
    v8h ra0 = *(const v8h*)gA0;
    v8h ra1 = *(const v8h*)gA1;
    v8h rb0 = *(const v8h*)gB0;
    v8h rb1 = *(const v8h*)gB1;

    for (int kt = 0; kt < K; kt += 32) {
        __syncthreads();               // previous tile's LDS reads complete
        *(v8h*)sA0 = ra0;
        *(v8h*)sA1 = ra1;
        *(v8h*)sB0 = rb0;
        *(v8h*)sB1 = rb1;
        __syncthreads();               // tile visible to all waves

        if (kt + 32 < K) {             // issue next tile's global loads under the WMMAs
            ra0 = *(const v8h*)(gA0 + kt + 32);
            ra1 = *(const v8h*)(gA1 + kt + 32);
            rb0 = *(const v8h*)(gB0 + kt + 32);
            rb1 = *(const v8h*)(gB1 + kt + 32);
            if (kt + 64 < K) {         // prefetch one more tile ahead (global_prefetch_b8)
                __builtin_prefetch(gA0 + kt + 64, 0, 0);
                __builtin_prefetch(gB0 + kt + 64, 0, 0);
            }
        }

        v16h a0 = ld16_lds(fA0, fA0 + 16);
        v16h a1 = ld16_lds(fA1, fA1 + 16);
#pragma unroll
        for (int j = 0; j < 4; ++j) {
            const _Float16* fb = fB0 + j * 16 * LDS_STRIDE;
            v16h b = ld16_lds(fb, fb + 16);
            c[0][j] = wmma16(a0, b, c[0][j]);
            c[1][j] = wmma16(a1, b, c[1][j]);
        }
    }

    // ---- epilogue
#pragma unroll
    for (int ms = 0; ms < 2; ++ms) {
#pragma unroll
        for (int r = 0; r < 8; ++r) {
            const int m = m0 + wr * 32 + ms * 16 + r + 8 * hs;
#pragma unroll
            for (int j = 0; j < 4; ++j) {
                const int n = n0 + wc * 64 + j * 16 + col;
                float v = c[ms][j][r];
                if (MODE == 2) {
                    if (bias) v += bias[n];
                    v += res[(size_t)m * N + n];
                    ((float*)outp)[(size_t)m * N + n] = v;
                } else if (MODE == 3) {
                    v = fmaxf(v + bias[n], 0.f);
                    ((half_t*)outp)[(size_t)m * N + n] = (half_t)v;
                } else if (MODE == 1) {
                    // scatter into Vt[(b*H+h)*DV+dv][S]
                    const int bb = m >> 11;          // m / S
                    const int s  = m & (SS - 1);
                    const int hh = n >> 6, dv = n & 63;
                    ((half_t*)outp)[((size_t)((bb * HH + hh) * DKV + dv)) * SS + s] = (half_t)v;
                } else {
                    ((half_t*)outp)[(size_t)m * N + n] = (half_t)v;
                }
            }
        }
    }
}

// ---------- flash attention: one wave per 16-row Q tile ----------
// Q,K: f16 [b,s,h,dk] row-major; Vt: f16 [(b*H+h)*DV+dv][S]; bias f32 [B,1,S,S]
// O: f16 [b,s,h,dv]
__global__ __launch_bounds__(128)
void flash_attn_kernel(const half_t* __restrict__ Q, const half_t* __restrict__ Km,
                       const half_t* __restrict__ Vt, const float* __restrict__ bias,
                       half_t* __restrict__ O) {
    __shared__ _Float16 lds_p[4][16 * 32];   // per-wave P staging (C-layout -> A-layout)

    const int lane = threadIdx.x & 31;
    const int wave = threadIdx.x >> 5;
    const int tile = blockIdx.x * 4 + wave;  // 0 .. B*H*(S/16)-1
    const int qt = tile & 127;
    const int h  = (tile >> 7) & (HH - 1);
    const int b  = tile >> 11;
    const int q0 = qt * 16;
    const int col = lane & 15;
    const int hs  = lane >> 4;
    const int kb  = hs * 8;

    // Q fragments (contraction over DK=64 -> two k-steps of 32)
    const half_t* qbase = Q + ((size_t)(b * SS + q0 + col) * DD + h * DKV);
    const v16h aq0 = ld16(qbase + kb,      qbase + kb + 16);
    const v16h aq1 = ld16(qbase + 32 + kb, qbase + 32 + kb + 16);

    float m_run[8], l_run[8];
    v8f oc[4] = {{}, {}, {}, {}};
#pragma unroll
    for (int r = 0; r < 8; ++r) { m_run[r] = -1e30f; l_run[r] = 0.f; }

    const size_t bias_row0 = ((size_t)b * SS + q0) * SS;

    for (int kt = 0; kt < SS; kt += 32) {
        // ---- S tiles: s0 covers keys kt..kt+15, s1 covers kt+16..kt+31 ----
        v8f s0 = {}, s1 = {};
        const half_t* kb0 = Km + ((size_t)(b * SS + kt + col) * DD + h * DKV);
        const half_t* kb1 = kb0 + (size_t)16 * DD;
        v16h bk;
        bk = ld16(kb0 + kb,      kb0 + kb + 16);      s0 = wmma16(aq0, bk, s0);
        bk = ld16(kb0 + 32 + kb, kb0 + 32 + kb + 16); s0 = wmma16(aq1, bk, s0);
        bk = ld16(kb1 + kb,      kb1 + kb + 16);      s1 = wmma16(aq0, bk, s1);
        bk = ld16(kb1 + 32 + kb, kb1 + 32 + kb + 16); s1 = wmma16(aq1, bk, s1);

        // ---- scale (D^-0.5 = 1/32) + bias, online softmax ----
        float p0[8], p1[8];
#pragma unroll
        for (int r = 0; r < 8; ++r) {
            const size_t brow = bias_row0 + (size_t)(r + 8 * hs) * SS + kt;
            const float sv0 = s0[r] * 0.03125f + bias[brow + col];
            const float sv1 = s1[r] * 0.03125f + bias[brow + 16 + col];
            float mx = redmax16(fmaxf(sv0, sv1));
            const float mn = fmaxf(m_run[r], mx);
            const float alpha = __expf(m_run[r] - mn);
            m_run[r] = mn;
            const float e0 = __expf(sv0 - mn);
            const float e1 = __expf(sv1 - mn);
            l_run[r] = l_run[r] * alpha + redsum16(e0 + e1);
#pragma unroll
            for (int j = 0; j < 4; ++j) oc[j][r] *= alpha;
            p0[r] = e0; p1[r] = e1;
        }

        // ---- P: C-layout f32 -> LDS -> A-layout f16 ----
        _Float16* lp = lds_p[wave];
#pragma unroll
        for (int r = 0; r < 8; ++r) {
            lp[(r + 8 * hs) * 32 + col]      = (_Float16)p0[r];
            lp[(r + 8 * hs) * 32 + 16 + col] = (_Float16)p1[r];
        }
        asm volatile("s_wait_dscnt 0" ::: "memory");
        const v16h ap = ld16_lds(lp + col * 32 + kb, lp + col * 32 + kb + 16);

        // ---- O += P x V (contraction over 32 keys, 4 dv tiles) ----
        const half_t* vb = Vt + ((size_t)((b * HH + h) * DKV + col)) * SS + kt + kb;
#pragma unroll
        for (int j = 0; j < 4; ++j) {
            const half_t* vp = vb + (size_t)(j * 16) * SS;
            v16h bv = ld16(vp, vp + 16);
            oc[j] = wmma16(ap, bv, oc[j]);
        }
    }

    // ---- normalize and store ----
#pragma unroll
    for (int r = 0; r < 8; ++r) {
        const float inv = 1.0f / l_run[r];
        const size_t obase = ((size_t)(b * SS + q0 + r + 8 * hs)) * DD + h * DKV;
#pragma unroll
        for (int j = 0; j < 4; ++j)
            O[obase + j * 16 + col] = (half_t)(oc[j][r] * inv);
    }
}

// ---------- orchestration ----------
extern "C" void kernel_launch(void* const* d_in, const int* in_sizes, int n_in,
                              void* d_out, int out_size, void* d_ws, size_t ws_size,
                              hipStream_t stream) {
    const float* enc   = (const float*)d_in[0];
    const float* abias = (const float*)d_in[1];
    const float* wq    = (const float*)d_in[2];
    const float* wk    = (const float*)d_in[3];
    const float* wv    = (const float*)d_in[4];
    const float* wo    = (const float*)d_in[5];
    const float* ln1g  = (const float*)d_in[6];
    const float* ln1b  = (const float*)d_in[7];
    const float* ln2g  = (const float*)d_in[8];
    const float* ln2b  = (const float*)d_in[9];
    const float* w1    = (const float*)d_in[10];
    const float* b1    = (const float*)d_in[11];
    const float* w2    = (const float*)d_in[12];
    const float* b2    = (const float*)d_in[13];
    float* out = (float*)d_out;

    char* ws = (char*)d_ws;
    auto carve = [&](size_t bytes) { char* p = ws; ws += bytes; return p; };
    half_t* x_h   = (half_t*)carve((size_t)MROWS * DD * 2);       // 8 MB
    half_t* wqT   = (half_t*)carve((size_t)DD * DD * 2);          // 2 MB
    half_t* wkT   = (half_t*)carve((size_t)DD * DD * 2);
    half_t* wvT   = (half_t*)carve((size_t)DD * DD * 2);
    half_t* woT   = (half_t*)carve((size_t)DD * DD * 2);
    half_t* w1T   = (half_t*)carve((size_t)DD * DFF_ * 2);        // 8 MB
    half_t* w2T   = (half_t*)carve((size_t)DD * DFF_ * 2);        // 8 MB
    half_t* q_h   = (half_t*)carve((size_t)MROWS * DD * 2);       // 8 MB
    half_t* k_h   = (half_t*)carve((size_t)MROWS * DD * 2);
    half_t* vt_h  = (half_t*)carve((size_t)MROWS * DD * 2);
    half_t* at_h  = (half_t*)carve((size_t)MROWS * DD * 2);
    float*  aout  = (float*)carve((size_t)MROWS * DD * 4);        // 16 MB
    half_t* y_h   = (half_t*)carve((size_t)MROWS * DD * 2);
    half_t* h_h   = (half_t*)carve((size_t)MROWS * DFF_ * 2);     // 32 MB

    // 1) weight transpose + cast to f16
    transpose_cast_kernel<<<(DD * DD) / 256, 256, 0, stream>>>(wq, wqT, DD, DD);
    transpose_cast_kernel<<<(DD * DD) / 256, 256, 0, stream>>>(wk, wkT, DD, DD);
    transpose_cast_kernel<<<(DD * DD) / 256, 256, 0, stream>>>(wv, wvT, DD, DD);
    transpose_cast_kernel<<<(DD * DD) / 256, 256, 0, stream>>>(wo, woT, DD, DD);
    transpose_cast_kernel<<<(DD * DFF_) / 256, 256, 0, stream>>>(w1, w1T, DD, DFF_);
    transpose_cast_kernel<<<(DD * DFF_) / 256, 256, 0, stream>>>(w2, w2T, DFF_, DD);

    // 2) LN1
    layernorm_f16_kernel<<<MROWS, 256, 0, stream>>>(enc, ln1g, ln1b, x_h);

    // 3) Q/K/V projections (block tile 128x128)
    dim3 gQ(DD / 128, MROWS / 128);
    gemm_wmma<0><<<gQ, 256, 0, stream>>>(x_h, wqT, nullptr, nullptr, q_h, MROWS, DD, DD);
    gemm_wmma<0><<<gQ, 256, 0, stream>>>(x_h, wkT, nullptr, nullptr, k_h, MROWS, DD, DD);
    gemm_wmma<1><<<gQ, 256, 0, stream>>>(x_h, wvT, nullptr, nullptr, vt_h, MROWS, DD, DD);

    // 4) flash attention (B*H*(S/16) = 4096 wave-tiles, 4 waves/block)
    flash_attn_kernel<<<(BB * HH * (SS / 16)) / 4, 128, 0, stream>>>(q_h, k_h, vt_h, abias, at_h);

    // 5) output projection + residual -> f32
    gemm_wmma<2><<<gQ, 256, 0, stream>>>(at_h, woT, nullptr, enc, aout, MROWS, DD, DD);

    // 6) LN2
    layernorm_f16_kernel<<<MROWS, 256, 0, stream>>>(aout, ln2g, ln2b, y_h);

    // 7) FFN
    dim3 gF1(DFF_ / 128, MROWS / 128);
    gemm_wmma<3><<<gF1, 256, 0, stream>>>(y_h, w1T, b1, nullptr, h_h, MROWS, DFF_, DD);
    gemm_wmma<2><<<gQ, 256, 0, stream>>>(h_h, w2T, b2, aout, out, MROWS, DD, DFF_);
}
